// GNNLanguageModel_65687229825555
// MI455X (gfx1250) — matrix operand order repaired and checked by reading, not compiled, not run on hardware
//
#include <hip/hip_runtime.h>

// ---------------------------------------------------------------- constants
#define NB     2
#define SEQ    2048
#define DMODEL 512
#define VOC    32000
#define NHEAD  4
#define WIN    3
#define DH     128           // DMODEL / NHEAD
#define NN     4096          // NB * SEQ

// ---------------------------------------------------------------- bf16 utils
typedef __attribute__((ext_vector_type(16))) __bf16 v16bf;
typedef __attribute__((ext_vector_type(8)))  float  v8f;
typedef __attribute__((ext_vector_type(4)))  int    v4i;

__device__ __forceinline__ unsigned int f2bf_bits(float f) {
    unsigned int u = __float_as_uint(f);
    u += 0x7FFFu + ((u >> 16) & 1u);          // round-to-nearest-even
    return u >> 16;
}
__device__ __forceinline__ unsigned short f2bf(float f) {
    return (unsigned short)f2bf_bits(f);
}
__device__ __forceinline__ unsigned int pack_bf2(float lo, float hi) {
    return f2bf_bits(lo) | (f2bf_bits(hi) << 16);
}

union FragU { uint4 q[2]; v16bf v; };

__device__ __forceinline__ v8f wmma_bf16(v16bf a, v16bf b, v8f c) {
    return __builtin_amdgcn_wmma_f32_16x16x32_bf16(
        /*neg_a=*/false, a, /*neg_b=*/false, b,
        /*c_mod=*/(short)0, c, /*reuse_a=*/false, /*reuse_b=*/false);
}

// async global->LDS path (guarded: falls back to VGPR staging if absent)
#if defined(__has_builtin)
#if __has_builtin(__builtin_amdgcn_global_load_async_to_lds_b128) && \
    __has_builtin(__builtin_amdgcn_s_wait_asynccnt)
#define HAVE_ASYNC_LDS 1
#endif
#endif
#ifndef HAVE_ASYNC_LDS
#define HAVE_ASYNC_LDS 0
#endif

#if HAVE_ASYNC_LDS
typedef __attribute__((address_space(1))) v4i GV4;   // global int4*
typedef __attribute__((address_space(3))) v4i LV4;   // LDS int4*
__device__ __forceinline__ void async_b128(const void* g, void* l) {
    __builtin_amdgcn_global_load_async_to_lds_b128(
        (GV4*)g, (LV4*)l, 0, 0);
}
#endif

// ---------------------------------------------------------------- fp32 -> bf16 convert (x4)
__global__ __launch_bounds__(256) void k_cvt4(const float* __restrict__ w,
                                              unsigned int* __restrict__ o, int count4) {
    int i = blockIdx.x * 256 + threadIdx.x;
    if (i < count4) {
        const float4 v = ((const float4*)w)[i];
        uint2 r;
        r.x = pack_bf2(v.x, v.y);
        r.y = pack_bf2(v.z, v.w);
        ((uint2*)o)[i] = r;
    }
}

// ---------------------------------------------------------------- embedding gather
__global__ __launch_bounds__(256) void k_embed(const int* __restrict__ ids,
                                               const float* __restrict__ emb,
                                               unsigned int* __restrict__ xb) {
    const int n   = blockIdx.x;
    const int row = ids[n];
    const float2* e = (const float2*)(emb + (size_t)row * DMODEL);
    unsigned int* o = xb + (size_t)n * (DMODEL / 2);
    const int d = threadIdx.x;            // 256 threads x 2 elems = 512
    const float2 v = e[d];
    o[d] = pack_bf2(v.x, v.y);
}

// ---------------------------------------------------------------- WMMA GEMM  C = A @ B^T
// A: [M,K] bf16 row-major, B: [Ncol,K] bf16 row-major (x @ W.T layout).
// Block tile 128x128, BK=32, 256 threads = 8 waves, wave tile 32x64 (2x4 frags).
// Double-buffered LDS, one barrier per K-step; async direct-to-LDS when available.
#define LDS_STRIDE 40   // 32 + 8 bf16 pad (16B) per row

__device__ __forceinline__ v16bf frag_a(const unsigned short* s, int row0) {
    const int lane = threadIdx.x & 31;
    const unsigned short* p = s + (row0 + (lane & 15)) * LDS_STRIDE + ((lane >> 4) << 3);
    FragU u;
    u.q[0] = *(const uint4*)(p);        // K = kg+0..7   (kg = 0 or 8)
    u.q[1] = *(const uint4*)(p + 16);   // K = kg+16..23
    return u.v;
}
__device__ __forceinline__ v16bf frag_b(const unsigned short* s, int col0) {
    const int lane = threadIdx.x & 31;
    const unsigned short* p = s + (col0 + (lane & 15)) * LDS_STRIDE + ((lane >> 4) << 3);
    FragU u;
    u.q[0] = *(const uint4*)(p);        // K = kg+0..7   (kg = 0 or 8)
    u.q[1] = *(const uint4*)(p + 16);   // K = kg+16..23
    return u.v;
}

template<bool BIAS, bool ELU_ACT, bool OUTBF>
__global__ __launch_bounds__(256) void k_gemm(const unsigned short* __restrict__ A,
                                              const unsigned short* __restrict__ Bw,
                                              const float* __restrict__ bias,
                                              float* __restrict__ Cf,
                                              unsigned short* __restrict__ Cb,
                                              int M, int Ncol, int K) {
    __shared__ unsigned short As[2][128 * LDS_STRIDE];
    __shared__ unsigned short Bs[2][128 * LDS_STRIDE];

    const int t    = threadIdx.x;
    const int lane = t & 31;
    const int wid  = t >> 5;
    const int wm   = (wid >> 1) * 32;   // wave row offset inside tile (0..96)
    const int wn   = (wid & 1) * 64;    // wave col offset inside tile (0|64)
    const int rowBase = blockIdx.y * 128;
    const int colBase = blockIdx.x * 128;

    // staging coords: each thread moves 2 x uint4 per 128x32 tile
    const int r0s = t >> 2;             // rows 0..63
    const int r1s = r0s + 64;           // rows 64..127
    const int cs  = (t & 3) << 3;       // bf16 col offset: 0,8,16,24
    const size_t aOff0 = (size_t)(rowBase + r0s) * K + cs;
    const size_t aOff1 = (size_t)(rowBase + r1s) * K + cs;
    const size_t bOff0 = (size_t)(colBase + r0s) * K + cs;
    const size_t bOff1 = (size_t)(colBase + r1s) * K + cs;
    const int l0 = r0s * LDS_STRIDE + cs;
    const int l1 = r1s * LDS_STRIDE + cs;

    v8f acc[2][4] = {};

    auto compute = [&](int b) {
        v16bf af[2], bfr[4];
        #pragma unroll
        for (int i = 0; i < 2; ++i) af[i] = frag_a(As[b], wm + i * 16);
        #pragma unroll
        for (int j = 0; j < 4; ++j) bfr[j] = frag_b(Bs[b], wn + j * 16);
        #pragma unroll
        for (int i = 0; i < 2; ++i)
            #pragma unroll
            for (int j = 0; j < 4; ++j)
                acc[i][j] = wmma_bf16(af[i], bfr[j], acc[i][j]);
    };

    int buf = 0;
#if HAVE_ASYNC_LDS
    auto stage = [&](int b, int kb) {
        async_b128(A  + aOff0 + kb, &As[b][l0]);
        async_b128(A  + aOff1 + kb, &As[b][l1]);
        async_b128(Bw + bOff0 + kb, &Bs[b][l0]);
        async_b128(Bw + bOff1 + kb, &Bs[b][l1]);
    };
    stage(0, 0);
    __builtin_amdgcn_s_wait_asynccnt(0);
    __syncthreads();
    for (int kb = 0; kb < K; kb += 32) {
        if (kb + 32 < K) stage(buf ^ 1, kb + 32);   // prefetch next tile into other buffer
        compute(buf);
        __builtin_amdgcn_s_wait_asynccnt(0);        // next tile landed (overlapped with WMMA)
        __syncthreads();
        buf ^= 1;
    }
#else
    uint4 ra0, ra1, rb0, rb1;
    auto gload = [&](int kb) {
        ra0 = *(const uint4*)(A  + aOff0 + kb);
        ra1 = *(const uint4*)(A  + aOff1 + kb);
        rb0 = *(const uint4*)(Bw + bOff0 + kb);
        rb1 = *(const uint4*)(Bw + bOff1 + kb);
    };
    auto commit = [&](int b) {
        *(uint4*)&As[b][l0] = ra0;
        *(uint4*)&As[b][l1] = ra1;
        *(uint4*)&Bs[b][l0] = rb0;
        *(uint4*)&Bs[b][l1] = rb1;
    };
    gload(0);
    commit(0);
    __syncthreads();
    for (int kb = 0; kb < K; kb += 32) {
        const bool hasNext = (kb + 32) < K;
        if (hasNext) gload(kb + 32);    // global loads overlap WMMA on current buffer
        compute(buf);
        if (hasNext) commit(buf ^ 1);   // other buffer: safe while buf is being read
        __syncthreads();
        buf ^= 1;
    }
#endif

    // store: C/D layout -> VGPR r, lane l : row = r + 8*(l>>4), col = l&15
    #pragma unroll
    for (int i = 0; i < 2; ++i) {
        #pragma unroll
        for (int j = 0; j < 4; ++j) {
            const int r0 = rowBase + wm + i * 16 + ((lane >> 4) << 3);
            const int c0 = colBase + wn + j * 16 + (lane & 15);
            float bv = 0.f;
            if (BIAS) bv = bias[c0];
            #pragma unroll
            for (int r = 0; r < 8; ++r) {
                float v = acc[i][j][r];
                if (BIAS) v += bv;
                if (ELU_ACT) v = (v > 0.f) ? v : (__expf(v) - 1.f);
                if (OUTBF) Cb[(size_t)(r0 + r) * Ncol + c0] = f2bf(v);
                else       Cf[(size_t)(r0 + r) * Ncol + c0] = v;
            }
        }
    }
}

// ---------------------------------------------------------------- GAT scores
// s1[n,h] = <Wx[n,h,:], a[h,0:DH]> ; s2[n,h] = <Wx[n,h,:], a[h,DH:2DH]>
__global__ __launch_bounds__(256) void k_scores(const float* __restrict__ Wx,
                                                const float* __restrict__ a,
                                                float* __restrict__ s1,
                                                float* __restrict__ s2) {
    const int gw   = (blockIdx.x * 256 + threadIdx.x) >> 5;  // (n,h) index
    const int lane = threadIdx.x & 31;
    const int n = gw >> 2, h = gw & 3;
    const float* wr = Wx + (size_t)n * DMODEL + h * DH;
    const float* ah = a + (size_t)h * (2 * DH);
    float p1 = 0.f, p2 = 0.f;
    #pragma unroll
    for (int i = lane; i < DH; i += 32) {
        const float v = wr[i];
        p1 += v * ah[i];
        p2 += v * ah[DH + i];
    }
    #pragma unroll
    for (int off = 16; off > 0; off >>= 1) {
        p1 += __shfl_xor(p1, off);
        p2 += __shfl_xor(p2, off);
    }
    if (lane == 0) { s1[gw] = p1; s2[gw] = p2; }
}

// ---------------------------------------------------------------- window softmax + aggregation
// Node j's incoming edges are exactly src = j+o, o in [-W,W]\{0} within the sequence;
// e(i->j) = lrelu(s1[i] + s2[j]); softmax over o; agg = sum alpha_o * Wx[src].
__global__ __launch_bounds__(256) void k_aggregate(const float* __restrict__ Wx,
                                                   const float* __restrict__ s1,
                                                   const float* __restrict__ s2,
                                                   unsigned short* __restrict__ aggb) {
    const int gw   = (blockIdx.x * 256 + threadIdx.x) >> 5;  // (n,h)
    const int lane = threadIdx.x & 31;
    const int n = gw >> 2, h = gw & 3;
    const int pp = n & (SEQ - 1);
    const float sd = s2[gw];

    float e[2 * WIN];
    int   srcn[2 * WIN];
    int ne = 0;
    #pragma unroll
    for (int o = -WIN; o <= WIN; ++o) {
        if (o == 0) continue;
        const int q = pp + o;
        if (q >= 0 && q < SEQ) {
            const int m = n + o;
            float ev = s1[m * NHEAD + h] + sd;
            e[ne]    = (ev >= 0.f) ? ev : 0.2f * ev;   // LeakyReLU(0.2)
            srcn[ne] = m;
            ++ne;
        }
    }
    float mx = -3.4e38f;
    for (int i = 0; i < ne; ++i) mx = fmaxf(mx, e[i]);
    float den = 1e-16f;
    for (int i = 0; i < ne; ++i) { e[i] = __expf(e[i] - mx); den += e[i]; }
    const float inv = 1.f / den;

    const int d0 = lane << 2;   // 4 floats/lane covers DH=128
    float a0 = 0.f, a1 = 0.f, a2 = 0.f, a3 = 0.f;
    for (int i = 0; i < ne; ++i) {
        const float w = e[i] * inv;
        const float4 v = *(const float4*)(Wx + (size_t)srcn[i] * DMODEL + h * DH + d0);
        a0 += w * v.x; a1 += w * v.y; a2 += w * v.z; a3 += w * v.w;
    }
    unsigned int* op = (unsigned int*)(aggb + (size_t)n * DMODEL + h * DH + d0);
    op[0] = pack_bf2(a0, a1);
    op[1] = pack_bf2(a2, a3);
}

// ---------------------------------------------------------------- launch
extern "C" void kernel_launch(void* const* d_in, const int* in_sizes, int n_in,
                              void* d_out, int out_size, void* d_ws, size_t ws_size,
                              hipStream_t stream) {
    const int*   input_ids = (const int*)  d_in[0];
    const float* emb       = (const float*)d_in[1];
    const float* W1        = (const float*)d_in[2];
    const float* a1        = (const float*)d_in[3];
    const float* o1_w      = (const float*)d_in[4];
    const float* o1_b      = (const float*)d_in[5];
    const float* W2        = (const float*)d_in[6];
    const float* a2        = (const float*)d_in[7];
    const float* o2_w      = (const float*)d_in[8];
    const float* o2_b      = (const float*)d_in[9];
    const float* lm_w      = (const float*)d_in[10];
    const float* lm_b      = (const float*)d_in[11];
    // d_in[12]/d_in[13] (src/dst) unused: window structure is closed-form.
    float* out = (float*)d_out;

    // workspace carve-up (256B aligned)
    char* p = (char*)d_ws;
    auto carve = [&](size_t bytes) -> void* {
        void* r = (void*)p;
        p += (bytes + 255) & ~(size_t)255;
        return r;
    };
    unsigned short* xbA  = (unsigned short*)carve((size_t)NN * DMODEL * 2);
    unsigned short* xbB  = (unsigned short*)carve((size_t)NN * DMODEL * 2);
    unsigned short* aggb = (unsigned short*)carve((size_t)NN * DMODEL * 2);
    float*          Wxf  = (float*)         carve((size_t)NN * DMODEL * 4);
    float*          s1   = (float*)         carve((size_t)NN * NHEAD * 4);
    float*          s2   = (float*)         carve((size_t)NN * NHEAD * 4);
    unsigned short* wb1  = (unsigned short*)carve((size_t)DMODEL * DMODEL * 2);
    unsigned short* wob1 = (unsigned short*)carve((size_t)DMODEL * DMODEL * 2);
    unsigned short* wb2  = (unsigned short*)carve((size_t)DMODEL * DMODEL * 2);
    unsigned short* wob2 = (unsigned short*)carve((size_t)DMODEL * DMODEL * 2);
    unsigned short* lmwb = (unsigned short*)carve((size_t)VOC * DMODEL * 2);

    const int DD4  = DMODEL * DMODEL / 4;    // 65536 float4s
    const int LMN4 = VOC * DMODEL / 4;       // 4096000 float4s

    // weights -> bf16 (vectorized: float4 in, packed 4xbf16 out)
    k_cvt4<<<(DD4  + 255) / 256, 256, 0, stream>>>(W1,   (unsigned int*)wb1,  DD4);
    k_cvt4<<<(DD4  + 255) / 256, 256, 0, stream>>>(o1_w, (unsigned int*)wob1, DD4);
    k_cvt4<<<(DD4  + 255) / 256, 256, 0, stream>>>(W2,   (unsigned int*)wb2,  DD4);
    k_cvt4<<<(DD4  + 255) / 256, 256, 0, stream>>>(o2_w, (unsigned int*)wob2, DD4);
    k_cvt4<<<(LMN4 + 255) / 256, 256, 0, stream>>>(lm_w, (unsigned int*)lmwb, LMN4);

    // embedding gather -> xbA (bf16)
    k_embed<<<NN, 256, 0, stream>>>(input_ids, emb, (unsigned int*)xbA);

    const dim3 gSmall(DMODEL / 128, NN / 128);   // (4, 32)
    const dim3 gLM(VOC / 128, NN / 128);         // (250, 32)
    const int  gNH = (NN * NHEAD * 32) / 256;    // 2048 blocks, wave per (n,h)

    // ---------------- GAT layer 1
    k_gemm<false, false, false><<<gSmall, 256, 0, stream>>>(
        xbA, wb1, nullptr, Wxf, nullptr, NN, DMODEL, DMODEL);
    k_scores<<<gNH, 256, 0, stream>>>(Wxf, a1, s1, s2);
    k_aggregate<<<gNH, 256, 0, stream>>>(Wxf, s1, s2, aggb);
    k_gemm<true, true, true><<<gSmall, 256, 0, stream>>>(
        aggb, wob1, o1_b, nullptr, xbB, NN, DMODEL, DMODEL);

    // ---------------- GAT layer 2
    k_gemm<false, false, false><<<gSmall, 256, 0, stream>>>(
        xbB, wb2, nullptr, Wxf, nullptr, NN, DMODEL, DMODEL);
    k_scores<<<gNH, 256, 0, stream>>>(Wxf, a2, s1, s2);
    k_aggregate<<<gNH, 256, 0, stream>>>(Wxf, s1, s2, aggb);
    k_gemm<true, true, true><<<gSmall, 256, 0, stream>>>(
        aggb, wob2, o2_b, nullptr, xbA, NN, DMODEL, DMODEL);

    // ---------------- LM head: logits[N, V] = x @ lm_w.T + lm_b
    k_gemm<true, false, false><<<gLM, 256, 0, stream>>>(
        xbA, lmwb, lm_b, out, nullptr, NN, VOC, DMODEL);
}
